// Qwen3Attention_23227183137032
// MI455X (gfx1250) — compile-verified, matrix operand order checked by reading, not run on hardware
//
#include <hip/hip_runtime.h>
#include <stdint.h>

typedef __attribute__((ext_vector_type(16))) __bf16 v16bf;
typedef __attribute__((ext_vector_type(8)))  float  v8f;
typedef __attribute__((ext_vector_type(4)))  int    v4i;

#define T_SEQ 2048
#define HIDN  2048
#define NH    16
#define NKV   8
#define DH    128
#define QKVN  4096          // NH*DH + 2*NKV*DH
#define SCALE 0.08838834764831845f

#if __has_builtin(__builtin_amdgcn_global_load_async_to_lds_b128) && \
    __has_builtin(__builtin_amdgcn_s_wait_asynccnt)
#define USE_ASYNC_LDS 1
typedef __attribute__((address_space(1))) v4i* gv4i_p;
typedef __attribute__((address_space(3))) v4i* lv4i_p;
#define AS1(p) ((gv4i_p)(p))
#define AS3(p) ((lv4i_p)(p))
#endif

__device__ __forceinline__ __bf16 tobf(float f) { return (__bf16)f; }

// Load a 16x32 bf16 WMMA fragment (A-layout; also used for B when the
// source LDS tile is stored [n][k]).  ISA 7.12.2:
//   lanes 0-15: M=lane, VGPR j holds K-pair ((j>=4)?16:0)+(half?8:0)+2*(j&3)
// Each K-pair is contiguous in LDS -> one dword load per VGPR.
__device__ __forceinline__ v16bf load_frag(const __bf16* base, int row0,
                                           int stride, int kbase, int lane) {
  const int half = lane >> 4;
  const int m    = lane & 15;
  const __bf16* p = base + (row0 + m) * stride + kbase;
  union { v16bf v; uint32_t u[8]; } f;
#pragma unroll
  for (int j = 0; j < 8; ++j) {
    int k0 = ((j >= 4) ? 16 : 0) + (half ? 8 : 0) + (j & 3) * 2;
    f.u[j] = *(const uint32_t*)(p + k0);
  }
  return f.v;
}

__device__ __forceinline__ v8f wmma_bf16(v16bf a, v16bf b, v8f c) {
  return __builtin_amdgcn_wmma_f32_16x16x32_bf16(false, a, false, b,
                                                 (short)0, c, false, false);
}

// ---------------------------------------------------------------------------
// GEMM: C[M,N] (f32) = A[M,K] @ B[K,N],  A is f32 or bf16, B is f32.
// Block tile 128x128, K-stage 64, 256 threads = 8 waves (4x2 wave grid),
// each wave owns a 32x64 C tile = 2x4 WMMA accumulators.
// ---------------------------------------------------------------------------
template <bool A_IS_BF16>
__global__ __launch_bounds__(256) void gemm_bf16_kernel(const void* __restrict__ A_,
                                                        const float* __restrict__ B,
                                                        float* __restrict__ C,
                                                        int M, int N, int K) {
  __shared__ __bf16 As[128][68];
  __shared__ __bf16 Bs[128][68];   // stored [n][k]

  const int tid  = threadIdx.x;
  const int lane = tid & 31;
  const int w    = tid >> 5;
  const int wr   = w >> 1;         // 0..3
  const int wc   = w & 1;          // 0..1
  const int half = lane >> 4;
  const int bn0  = blockIdx.x * 128;
  const int bm0  = blockIdx.y * 128;

  v8f acc[2][4];
#pragma unroll
  for (int i = 0; i < 2; ++i)
#pragma unroll
    for (int t = 0; t < 4; ++t)
      acc[i][t] = (v8f){0.f, 0.f, 0.f, 0.f, 0.f, 0.f, 0.f, 0.f};

  for (int k0 = 0; k0 < K; k0 += 64) {
    // prefetch next stage into GL2 while this stage computes
    if (k0 + 64 < K) {
      const size_t esz = A_IS_BF16 ? 2 : 4;
      __builtin_prefetch((const char*)A_ +
                         ((size_t)(bm0 + (tid >> 1)) * K + k0 + 64) * esz, 0, 0);
      __builtin_prefetch(B + (size_t)(k0 + 64 + (tid >> 2)) * N + bn0 +
                         (tid & 3) * 32, 0, 0);
    }
    // stage A tile (128 x 64) -> bf16 LDS
#pragma unroll
    for (int i = 0; i < 8; ++i) {
      int idx = tid + 256 * i;
      int row = idx >> 4;
      int c4  = (idx & 15) * 4;
      if constexpr (A_IS_BF16) {
        const __bf16* Ab = (const __bf16*)A_;
        uint2 u = *(const uint2*)(Ab + (size_t)(bm0 + row) * K + k0 + c4);
        *(uint32_t*)&As[row][c4]     = u.x;
        *(uint32_t*)&As[row][c4 + 2] = u.y;
      } else {
        const float* Af = (const float*)A_;
        float4 v = *(const float4*)(Af + (size_t)(bm0 + row) * K + k0 + c4);
        As[row][c4]     = tobf(v.x);
        As[row][c4 + 1] = tobf(v.y);
        As[row][c4 + 2] = tobf(v.z);
        As[row][c4 + 3] = tobf(v.w);
      }
    }
    // stage B tile (64 x 128), transposed into Bs[n][k]
#pragma unroll
    for (int i = 0; i < 8; ++i) {
      int idx = tid + 256 * i;
      int kr  = idx >> 5;
      int c4  = (idx & 31) * 4;
      float4 v = *(const float4*)(B + (size_t)(k0 + kr) * N + bn0 + c4);
      Bs[c4][kr]     = tobf(v.x);
      Bs[c4 + 1][kr] = tobf(v.y);
      Bs[c4 + 2][kr] = tobf(v.z);
      Bs[c4 + 3][kr] = tobf(v.w);
    }
    __syncthreads();

#pragma unroll
    for (int kk = 0; kk < 64; kk += 32) {
      v16bf a0 = load_frag(&As[0][0], wr * 32,      68, kk, lane);
      v16bf a1 = load_frag(&As[0][0], wr * 32 + 16, 68, kk, lane);
#pragma unroll
      for (int t = 0; t < 4; ++t) {
        v16bf b = load_frag(&Bs[0][0], wc * 64 + 16 * t, 68, kk, lane);
        acc[0][t] = wmma_bf16(a0, b, acc[0][t]);
        acc[1][t] = wmma_bf16(a1, b, acc[1][t]);
      }
    }
    __syncthreads();
  }

  // epilogue: C layout row = r + 8*half, col = lane&15 within each 16x16 tile
#pragma unroll
  for (int i = 0; i < 2; ++i)
#pragma unroll
    for (int t = 0; t < 4; ++t)
#pragma unroll
      for (int r = 0; r < 8; ++r) {
        int row = bm0 + wr * 32 + 16 * i + r + 8 * half;
        int col = bn0 + wc * 64 + 16 * t + (lane & 15);
        C[(size_t)row * N + col] = acc[i][t][r];
      }
}

// ---------------------------------------------------------------------------
// RMSNorm + RoPE + scale, qkv f32 -> q/k/v bf16.
// grid (T, 4): part 0/1 = q heads 0-7 / 8-15, part 2 = k heads, part 3 = v.
// One wave per head row of 128; partner for rotate-half is in same lane.
// ---------------------------------------------------------------------------
__global__ __launch_bounds__(256) void rope_norm_kernel(const float* __restrict__ qkv,
                                                        const int* __restrict__ pos,
                                                        const float* __restrict__ qw,
                                                        const float* __restrict__ kw,
                                                        __bf16* __restrict__ qb,
                                                        __bf16* __restrict__ kb,
                                                        __bf16* __restrict__ vb) {
  const int t    = blockIdx.x;
  const int part = blockIdx.y;
  const int w    = threadIdx.x >> 5;
  const int lane = threadIdx.x & 31;

  const float* src;
  __bf16* dst;
  const float* nw = nullptr;
  float oscale = 1.0f;
  bool do_rope = true;

  if (part == 0) {
    src = qkv + (size_t)t * QKVN + w * DH;
    dst = qb + ((size_t)t * NH + w) * DH;
    nw = qw; oscale = SCALE;
  } else if (part == 1) {
    src = qkv + (size_t)t * QKVN + (8 + w) * DH;
    dst = qb + ((size_t)t * NH + 8 + w) * DH;
    nw = qw; oscale = SCALE;
  } else if (part == 2) {
    src = qkv + (size_t)t * QKVN + NH * DH + w * DH;
    dst = kb + ((size_t)t * NKV + w) * DH;
    nw = kw;
  } else {
    src = qkv + (size_t)t * QKVN + (NH + NKV) * DH + w * DH;
    dst = vb + ((size_t)t * NKV + w) * DH;
    do_rope = false;
  }

  float x[4];
#pragma unroll
  for (int j = 0; j < 4; ++j) x[j] = src[lane + 32 * j];

  if (!do_rope) {
#pragma unroll
    for (int j = 0; j < 4; ++j) dst[lane + 32 * j] = tobf(x[j]);
    return;
  }

  float ss = 0.f;
#pragma unroll
  for (int j = 0; j < 4; ++j) ss += x[j] * x[j];
#pragma unroll
  for (int m = 16; m >= 1; m >>= 1) ss += __shfl_xor(ss, m, 32);
  const float rinv = rsqrtf(ss * (1.0f / 128.0f) + 1e-6f);

  float xn[4];
#pragma unroll
  for (int j = 0; j < 4; ++j) xn[j] = x[j] * rinv * nw[lane + 32 * j];

  const float p = (float)pos[t];
#pragma unroll
  for (int j = 0; j < 4; ++j) {
    int d  = lane + 32 * j;
    int fi = d & 63;                                   // freq index
    // inv_freq = theta^(-fi/64) = exp(-fi * ln(10000)/64)
    float ang = p * __expf(-(float)fi * 0.14391156f);
    float cs, sn;
    __sincosf(ang, &sn, &cs);
    float rot = (j < 2) ? -xn[j + 2] : xn[j - 2];
    dst[d] = tobf((xn[j] * cs + rot * sn) * oscale);
  }
}

// ---------------------------------------------------------------------------
// Flash attention, causal, GQA.  grid (T/64, NH), 256 threads = 8 waves.
// BQ = 64 query rows, BKV = 32 key rows per stage.
// Q/K tiles staged with async global->LDS DMA (ASYNCcnt) when available.
// ---------------------------------------------------------------------------
__global__ __launch_bounds__(256) void flash_attn_kernel(const __bf16* __restrict__ qb,
                                                         const __bf16* __restrict__ kb,
                                                         const __bf16* __restrict__ vb,
                                                         __bf16* __restrict__ ob) {
  // stride 136 elems = 272 B = 17*16 B: B128-aligned rows, conflict-free frags
  __shared__ __bf16 Qs[64][136];
  __shared__ __bf16 Ks[32][136];
  __shared__ __bf16 Vst[128][36];   // V transposed: [d][kv]
  __shared__ float  Sls[64][33];
  __shared__ __bf16 Ps[64][36];
  __shared__ float  mRow[64], lRow[64], scRow[64];

  const int tid  = threadIdx.x;
  const int lane = tid & 31;
  const int w    = tid >> 5;
  const int half = lane >> 4;
  const int q0   = blockIdx.x * 64;
  const int h    = blockIdx.y;
  const int kh   = h >> 1;          // G = 2

  const int row0 = 16 * (w >> 1);
  const int scol = 16 * (w & 1);
  const int ocol = 64 * (w & 1);

  // ---- stage Q tile (64 x 128 bf16 = 1024 x 16B) ----
#ifdef USE_ASYNC_LDS
#pragma unroll
  for (int i = 0; i < 4; ++i) {
    int idx = tid + 256 * i;
    int row = idx >> 4;             // 16 x 16B chunks per row
    int c16 = (idx & 15) * 8;       // in bf16 elements
    const __bf16* g = qb + ((size_t)(q0 + row) * NH + h) * DH + c16;
    __builtin_amdgcn_global_load_async_to_lds_b128(AS1(g), AS3(&Qs[row][c16]),
                                                   0, 0);
  }
#else
#pragma unroll
  for (int i = 0; i < 16; ++i) {
    int idx = tid + 256 * i;
    int row = idx >> 6;
    int cu  = idx & 63;
    const uint32_t* s =
        (const uint32_t*)(qb + ((size_t)(q0 + row) * NH + h) * DH);
    *(uint32_t*)&Qs[row][2 * cu] = s[cu];
  }
#endif
  if (tid < 64) { mRow[tid] = -1e30f; lRow[tid] = 0.f; }

  v8f oacc[4];
#pragma unroll
  for (int t = 0; t < 4; ++t)
    oacc[t] = (v8f){0.f, 0.f, 0.f, 0.f, 0.f, 0.f, 0.f, 0.f};

#ifdef USE_ASYNC_LDS
  __builtin_amdgcn_s_wait_asynccnt(0);
#endif
  __syncthreads();

  for (int kbase = 0; kbase <= q0 + 63; kbase += 32) {
    // ---- stage K (32 x 128) via async DMA, V^T (128 x 32) via transpose ----
#ifdef USE_ASYNC_LDS
#pragma unroll
    for (int i = 0; i < 2; ++i) {
      int idx = tid + 256 * i;
      int row = idx >> 4;
      int c16 = (idx & 15) * 8;
      const __bf16* g = kb + ((size_t)(kbase + row) * NKV + kh) * DH + c16;
      __builtin_amdgcn_global_load_async_to_lds_b128(AS1(g), AS3(&Ks[row][c16]),
                                                     0, 0);
    }
#else
#pragma unroll
    for (int i = 0; i < 8; ++i) {
      int idx = tid + 256 * i;
      int row = idx >> 6;
      int cu  = idx & 63;
      const uint32_t* s =
          (const uint32_t*)(kb + ((size_t)(kbase + row) * NKV + kh) * DH);
      *(uint32_t*)&Ks[row][2 * cu] = s[cu];
    }
#endif
#pragma unroll
    for (int i = 0; i < 16; ++i) {
      int idx = tid + 256 * i;
      int row = idx >> 7;
      int d   = idx & 127;
      Vst[d][row] = vb[((size_t)(kbase + row) * NKV + kh) * DH + d];
    }
#ifdef USE_ASYNC_LDS
    __builtin_amdgcn_s_wait_asynccnt(0);
#endif
    __syncthreads();

    // S = Q @ K^T  (per-wave 16x16 tile, K-loop over D=128)
    v8f sacc = (v8f){0.f, 0.f, 0.f, 0.f, 0.f, 0.f, 0.f, 0.f};
#pragma unroll
    for (int kk = 0; kk < 128; kk += 32) {
      v16bf a = load_frag(&Qs[0][0], row0, 136, kk, lane);
      v16bf b = load_frag(&Ks[0][0], scol, 136, kk, lane);
      sacc = wmma_bf16(a, b, sacc);
    }
#pragma unroll
    for (int r = 0; r < 8; ++r)
      Sls[row0 + r + 8 * half][scol + (lane & 15)] = sacc[r];
    __syncthreads();

    // online softmax: 4 threads per row, 8 cols each
    {
      int r  = tid >> 2;
      int c0 = (tid & 3) * 8;
      float sv[8];
      float mx = -1e30f;
#pragma unroll
      for (int j = 0; j < 8; ++j) {
        int c = c0 + j;
        bool ok = (kbase + c) <= (q0 + r);
        sv[j] = ok ? Sls[r][c] : -1e30f;
        mx = fmaxf(mx, sv[j]);
      }
      mx = fmaxf(mx, __shfl_xor(mx, 1, 4));
      mx = fmaxf(mx, __shfl_xor(mx, 2, 4));
      float m_old = mRow[r];
      float m_new = fmaxf(m_old, mx);
      float sum = 0.f;
#pragma unroll
      for (int j = 0; j < 8; ++j) {
        float e = __expf(sv[j] - m_new);
        Ps[r][c0 + j] = tobf(e);
        sum += e;
      }
      sum += __shfl_xor(sum, 1, 4);
      sum += __shfl_xor(sum, 2, 4);
      if ((tid & 3) == 0) {
        float sc = __expf(m_old - m_new);
        scRow[r] = sc;
        lRow[r]  = lRow[r] * sc + sum;
        mRow[r]  = m_new;
      }
    }
    __syncthreads();

    // rescale O accumulators, then O += P @ V
    float sc[8];
#pragma unroll
    for (int r = 0; r < 8; ++r) sc[r] = scRow[row0 + r + 8 * half];
#pragma unroll
    for (int t = 0; t < 4; ++t)
#pragma unroll
      for (int r = 0; r < 8; ++r) oacc[t][r] *= sc[r];

    v16bf pa = load_frag(&Ps[0][0], row0, 36, 0, lane);
#pragma unroll
    for (int t = 0; t < 4; ++t) {
      v16bf bv = load_frag(&Vst[0][0], ocol + 16 * t, 36, 0, lane);
      oacc[t] = wmma_bf16(pa, bv, oacc[t]);
    }
    __syncthreads();
  }

  // normalize and write O (bf16) into o[T, H*D]
#pragma unroll
  for (int t = 0; t < 4; ++t)
#pragma unroll
    for (int r = 0; r < 8; ++r) {
      int row = row0 + r + 8 * half;
      int col = ocol + 16 * t + (lane & 15);
      float val = oacc[t][r] / lRow[row];
      ob[((size_t)(q0 + row) * NH + h) * DH + col] = tobf(val);
    }
}

// ---------------------------------------------------------------------------
extern "C" void kernel_launch(void* const* d_in, const int* in_sizes, int n_in,
                              void* d_out, int out_size, void* d_ws, size_t ws_size,
                              hipStream_t stream) {
  const float* x       = (const float*)d_in[0];
  const int*   pos     = (const int*)d_in[1];
  const float* w_qkv   = (const float*)d_in[2];
  const float* q_norm  = (const float*)d_in[3];
  const float* k_norm  = (const float*)d_in[4];
  const float* w_o     = (const float*)d_in[5];
  float*       out     = (float*)d_out;

  char* ws = (char*)d_ws;
  float*  qkv = (float*)ws;                                   // 32 MB
  __bf16* qb  = (__bf16*)(ws + (size_t)33554432);             //  8 MB
  __bf16* kb  = (__bf16*)(ws + (size_t)33554432 + 8388608);   //  4 MB
  __bf16* vb  = (__bf16*)(ws + (size_t)33554432 + 12582912);  //  4 MB
  __bf16* ob  = (__bf16*)(ws + (size_t)33554432 + 16777216);  //  8 MB

  // 1) qkv = x @ w_qkv   (2048 x 2048 x 4096)
  gemm_bf16_kernel<false><<<dim3(QKVN / 128, T_SEQ / 128), 256, 0, stream>>>(
      x, w_qkv, qkv, T_SEQ, QKVN, HIDN);

  // 2) RMSNorm + RoPE + scale -> bf16 q/k/v
  rope_norm_kernel<<<dim3(T_SEQ, 4), 256, 0, stream>>>(
      qkv, pos, q_norm, k_norm, qb, kb, vb);

  // 3) causal flash attention (GQA) -> o bf16
  flash_attn_kernel<<<dim3(T_SEQ / 64, NH), 256, 0, stream>>>(qb, kb, vb, ob);

  // 4) out = o @ w_o   (2048 x 2048 x 2048)
  gemm_bf16_kernel<true><<<dim3(HIDN / 128, T_SEQ / 128), 256, 0, stream>>>(
      ob, w_o, out, T_SEQ, HIDN, HIDN);
}
